// Layer_61907658605100
// MI455X (gfx1250) — matrix operand order repaired and checked by reading, not compiled, not run on hardware
//
#include <hip/hip_runtime.h>
#include <cstdint>
#include <cstddef>

// ---------------------------------------------------------------------------
// CDNA5 (gfx1250) transformer decoder layer.
// bf16 WMMA (16x16x32) everywhere; TDM (tensor_load_to_lds) double-buffered
// weight staging in the GEMMs; flash attention with LDS P-tile relayout.
// ---------------------------------------------------------------------------

typedef __attribute__((ext_vector_type(16))) __bf16 v16bf;
typedef __attribute__((ext_vector_type(8)))  float  v8f;
typedef __attribute__((ext_vector_type(4)))  unsigned int v4u;
typedef __attribute__((ext_vector_type(8)))  int v8i;
typedef __attribute__((ext_vector_type(4)))  int v4i;

#if __has_builtin(__builtin_amdgcn_tensor_load_to_lds)
#define ATHENA_HAVE_TDM 1
#else
#define ATHENA_HAVE_TDM 0
#endif

__device__ __forceinline__ unsigned short f2bf(float f) {
    unsigned int u = __float_as_uint(f);
    unsigned int r = u + 0x7FFFu + ((u >> 16) & 1u);
    return (unsigned short)(r >> 16);
}

// A-matrix fragment (16x32, MxK), 16-bit elements.
// lane L: row M=L%16, K in {k0+hi*8 .. +7} and {k0+16+hi*8 .. +7}
__device__ __forceinline__ v16bf load_a_frag(const unsigned short* row, int k0, int hi) {
    union { uint4 q[2]; v16bf v; } u;
    u.q[0] = *(const uint4*)(row + k0 + hi * 8);
    u.q[1] = *(const uint4*)(row + k0 + 16 + hi * 8);
    return u.v;
}

// B-matrix fragment (32x16, KxN), 16-bit elements.
// lane L: column N=L%16; K in {k0+hi*8 .. +7} and {k0+16+hi*8 .. +7}
__device__ __forceinline__ v16bf load_b_frag(const unsigned short* row, int k0, int hi) {
    union { uint4 q[2]; v16bf v; } u;
    const unsigned short* p = row + k0 + hi * 8;
    u.q[0] = *(const uint4*)(p);
    u.q[1] = *(const uint4*)(p + 16);
    return u.v;
}

__device__ __forceinline__ v8f wmma_bf16(v16bf a, v16bf b, v8f c) {
    return __builtin_amdgcn_wmma_f32_16x16x32_bf16(
        false, a, false, b, (short)0, c, false, false);
}

__device__ __forceinline__ v8f v8f_zero() {
    v8f z = {0.f, 0.f, 0.f, 0.f, 0.f, 0.f, 0.f, 0.f};
    return z;
}

#if ATHENA_HAVE_TDM
// Issue a 2D TDM load: tile (tile0 x tile1) of 2-byte elements from a
// row-major tensor (row length stride0 elements) at byte address ga, into
// LDS at byte offset ldsAddr (rows stored contiguously, tile0*2 bytes each).
// D# layout per CDNA5 ISA ch.8 (group0 128b, group1 256b).
// This toolchain exposes the 6-arg builtin:
//   (uint32x4 g0, int32x8 g1, int32x4 g2, int32x4 g3, int32x8 g4, i32 cpol)
__device__ __forceinline__ void tdm_load_2d(unsigned int ldsAddr,
                                            unsigned long long ga,
                                            int tile0, int tile1,
                                            int td0, int td1,
                                            long long stride0) {
    v4u g0;
    g0[0] = 1u;                                          // count=1, user mode
    g0[1] = ldsAddr;                                     // lds_addr (bytes)
    g0[2] = (unsigned)(ga & 0xFFFFFFFFull);              // global_addr[31:0]
    g0[3] = (unsigned)((ga >> 32) & 0x1FFFFFFull)        // global_addr[56:32]
          | (2u << 30);                                  // type = 2 (image)
    v8i g1;
    g1[0] = (int)(1u << 16);                             // data_size = 2 bytes
    g1[1] = (int)(((unsigned)td0 & 0xFFFFu) << 16);      // tensor_dim0[15:0]
    g1[2] = (int)((((unsigned)td0 >> 16) & 0xFFFFu)
          | (((unsigned)td1 & 0xFFFFu) << 16));          // td0[31:16] | td1[15:0]
    g1[3] = (int)((((unsigned)td1 >> 16) & 0xFFFFu)
          | (((unsigned)tile0 & 0xFFFFu) << 16));        // td1[31:16] | tile_dim0
    g1[4] = (int)((unsigned)tile1 & 0xFFFFu);            // tile_dim1, tile_dim2=0
    g1[5] = (int)((unsigned long long)stride0 & 0xFFFFFFFFull);  // stride0[31:0]
    g1[6] = (int)(((unsigned long long)stride0 >> 32) & 0xFFFFull); // stride0[47:32], stride1=0
    g1[7] = 0;
    const v4i z4 = {0, 0, 0, 0};
    const v8i z8 = {0, 0, 0, 0, 0, 0, 0, 0};
    __builtin_amdgcn_tensor_load_to_lds(g0, g1, z4, z4, z8, 0);
}
#endif

// ---------------------------------------------------------------------------
// fp32 -> bf16 cast
// ---------------------------------------------------------------------------
__global__ void cast_bf16_kernel(const float* __restrict__ in,
                                 unsigned short* __restrict__ out, int n) {
    int i = blockIdx.x * blockDim.x + threadIdx.x;
    if (i < n) out[i] = f2bf(in[i]);
}

// ---------------------------------------------------------------------------
// LayerNorm over D=1024, outputs bf16. One row per block (256 threads x 4).
// ---------------------------------------------------------------------------
__global__ __launch_bounds__(256)
void ln_bf16_kernel(const float* __restrict__ x, const float* __restrict__ g,
                    const float* __restrict__ bta, unsigned short* __restrict__ out,
                    int D) {
    __shared__ float s1[256];
    __shared__ float s2[256];
    const int row = blockIdx.x;
    const int tid = threadIdx.x;
    const float4 v = ((const float4*)(x + (size_t)row * D))[tid];
    float sum = v.x + v.y + v.z + v.w;
    float sq  = v.x * v.x + v.y * v.y + v.z * v.z + v.w * v.w;
    s1[tid] = sum; s2[tid] = sq;
    __syncthreads();
    for (int s = 128; s > 0; s >>= 1) {
        if (tid < s) { s1[tid] += s1[tid + s]; s2[tid] += s2[tid + s]; }
        __syncthreads();
    }
    const float mu   = s1[0] / (float)D;
    const float var  = s2[0] / (float)D - mu * mu;
    const float rstd = rsqrtf(var + 1e-5f);
    const int base = tid * 4;
    const float xs[4] = {v.x, v.y, v.z, v.w};
#pragma unroll
    for (int i = 0; i < 4; ++i) {
        const int c = base + i;
        out[(size_t)row * D + c] = f2bf(g[c] * (xs[i] - mu) * rstd + bta[c]);
    }
}

// ---------------------------------------------------------------------------
// GEMM: C[M,N] = A[M,K](bf16) @ W[N,K](bf16)^T  (+bias) (+relu)
// MODE 0: store bf16 row-major [M,N]
// MODE 1: fp32 out[M,N] += val   (residual accumulate)
// MODE 2: store bf16 [B,H,S,64]  (Q/K head layout), m=b*S+s, n=h*64+d
// MODE 3: store bf16 [B,H,64,S]  (V transposed)
//
// Block: 256 threads = 8 waves (4 m-waves x 2 n-waves); wave tile 32x64;
// block tile 128x128. Weight tile (128 rows x 32 k, 8KB) is staged into LDS
// by the Tensor Data Mover, double buffered; wave 0 issues the DMA, waits
// TENSORcnt, then the workgroup barrier publishes the tile. All waves read
// B fragments from LDS (ds_load_b128). A fragments stream from global/L2.
// ---------------------------------------------------------------------------
template <int MODE, bool RELU, bool HASBIAS>
__global__ __launch_bounds__(256)
void gemm_bf16_kernel(const unsigned short* __restrict__ A,
                      const unsigned short* __restrict__ W,
                      const float* __restrict__ bias,
                      void* __restrict__ outv,
                      int M, int N, int K, int S, int Hh) {
#if ATHENA_HAVE_TDM
    __shared__ unsigned short bTile[2][128 * 32];   // 2 x 8KB
#endif
    const int lane   = threadIdx.x & 31;
    const int wave   = threadIdx.x >> 5;
    const int laneLo = lane & 15;
    const int hi     = lane >> 4;
    const int mBase  = blockIdx.x * 128 + (wave >> 1) * 32;
    const int nBlock = blockIdx.y * 128;
    const int nBase  = nBlock + (wave & 1) * 64;

    v8f c[2][4];
#pragma unroll
    for (int it = 0; it < 2; ++it)
#pragma unroll
        for (int jt = 0; jt < 4; ++jt) c[it][jt] = v8f_zero();

    const unsigned short* aRow0 = A + (size_t)(mBase + laneLo) * K;
    const unsigned short* aRow1 = A + (size_t)(mBase + 16 + laneLo) * K;

    const int T = K >> 5;   // K / 32 k-steps

#if ATHENA_HAVE_TDM
    if (wave == 0) {
        tdm_load_2d((unsigned int)(uintptr_t)&bTile[0][0],
                    (unsigned long long)(uintptr_t)(W + (size_t)nBlock * K),
                    32, 128, K, N, (long long)K);
    }
#else
    const unsigned short* wRowBase = W + (size_t)(nBase + laneLo) * K;
#endif

    for (int kt = 0; kt < T; ++kt) {
        const int k = kt << 5;
#if ATHENA_HAVE_TDM
        if (wave == 0) {
            if (kt + 1 < T) {
                tdm_load_2d((unsigned int)(uintptr_t)&bTile[(kt + 1) & 1][0],
                            (unsigned long long)(uintptr_t)(W + (size_t)nBlock * K + k + 32),
                            32, 128, K, N, (long long)K);
                __builtin_amdgcn_s_wait_tensorcnt(1);
            } else {
                __builtin_amdgcn_s_wait_tensorcnt(0);
            }
        }
        __syncthreads();   // bTile[kt&1] now valid for all waves
        const unsigned short* bt = &bTile[kt & 1][0] + ((wave & 1) * 64) * 32;
#endif
        __builtin_prefetch(aRow0 + k + 64, 0, 1);
        const v16bf a0 = load_a_frag(aRow0, k, hi);
        const v16bf a1 = load_a_frag(aRow1, k, hi);
#pragma unroll
        for (int jt = 0; jt < 4; ++jt) {
#if ATHENA_HAVE_TDM
            const v16bf b = load_b_frag(bt + (jt * 16 + laneLo) * 32, 0, hi);
#else
            const v16bf b = load_b_frag(wRowBase + (size_t)(jt * 16) * K, k, hi);
#endif
            c[0][jt] = wmma_bf16(a0, b, c[0][jt]);
            c[1][jt] = wmma_bf16(a1, b, c[1][jt]);
        }
#if ATHENA_HAVE_TDM
        __syncthreads();   // everyone done reading before tile is overwritten
#endif
    }

#pragma unroll
    for (int it = 0; it < 2; ++it) {
#pragma unroll
        for (int jt = 0; jt < 4; ++jt) {
#pragma unroll
            for (int r = 0; r < 8; ++r) {
                const int m = mBase + it * 16 + r + 8 * hi;
                const int n = nBase + jt * 16 + laneLo;
                float val = c[it][jt][r];
                if (HASBIAS) val += bias[n];
                if (RELU) val = fmaxf(val, 0.f);
                if (MODE == 0) {
                    ((unsigned short*)outv)[(size_t)m * N + n] = f2bf(val);
                } else if (MODE == 1) {
                    ((float*)outv)[(size_t)m * N + n] += val;
                } else {
                    const int s = m % S;
                    const int b = m / S;
                    const int h = n >> 6;
                    const int d = n & 63;
                    if (MODE == 2) {
                        ((unsigned short*)outv)[(((size_t)(b * Hh + h)) * S + s) * 64 + d] = f2bf(val);
                    } else {
                        ((unsigned short*)outv)[(((size_t)(b * Hh + h)) * 64 + d) * S + s] = f2bf(val);
                    }
                }
            }
        }
    }
}

// ---------------------------------------------------------------------------
// Flash attention (dk = 64). Q,K: [B,H,S,64] bf16. Vt: [B,H,64,S] bf16.
// O: [B,S,H*64] bf16 (heads merged, ready for output projection).
// One wave handles 16 query rows; key tiles of 32; online softmax with
// 16-lane shfl_xor row reductions; P relayout via wave-private LDS tile.
// ---------------------------------------------------------------------------
__global__ __launch_bounds__(256)
void attn_kernel(const unsigned short* __restrict__ Q,
                 const unsigned short* __restrict__ Kh,
                 const unsigned short* __restrict__ Vt,
                 unsigned short* __restrict__ O,
                 int S, int Hh, float scale, int causal) {
    __shared__ unsigned short plds[8 * 16 * 32];  // per-wave 16x32 bf16 tile

    const int lane   = threadIdx.x & 31;
    const int wave   = threadIdx.x >> 5;
    const int laneLo = lane & 15;
    const int hi     = lane >> 4;
    const int bh     = blockIdx.x;
    const int b      = bh / Hh;
    const int h      = bh % Hh;
    const int D      = Hh * 64;
    const int qb     = blockIdx.y * 128 + wave * 16;
    unsigned short* myLds = plds + wave * 512;

    const unsigned short* qRow = Q + ((size_t)bh * S + qb + laneLo) * 64;
    const v16bf aq0 = load_a_frag(qRow, 0, hi);
    const v16bf aq1 = load_a_frag(qRow, 32, hi);

    float mrow[8], lrow[8];
    v8f o[4];
#pragma unroll
    for (int r = 0; r < 8; ++r) { mrow[r] = -3.0e38f; lrow[r] = 0.f; }
#pragma unroll
    for (int j = 0; j < 4; ++j) o[j] = v8f_zero();

    const int kEnd = causal ? (qb + 16) : S;
    for (int kb = 0; kb < kEnd; kb += 32) {
        const unsigned short* kRow0 = Kh + ((size_t)bh * S + kb + laneLo) * 64;
        const unsigned short* kRow1 = kRow0 + 16 * 64;

        v8f c0 = v8f_zero(), c1 = v8f_zero();
        c0 = wmma_bf16(aq0, load_b_frag(kRow0, 0, hi), c0);
        c0 = wmma_bf16(aq1, load_b_frag(kRow0, 32, hi), c0);
        c1 = wmma_bf16(aq0, load_b_frag(kRow1, 0, hi), c1);
        c1 = wmma_bf16(aq1, load_b_frag(kRow1, 32, hi), c1);

        const bool doMask = causal && (kb + 31 > qb);
#pragma unroll
        for (int r = 0; r < 8; ++r) {
            const int q = qb + r + 8 * hi;
            float v0 = c0[r] * scale;
            float v1 = c1[r] * scale;
            if (doMask) {
                if (kb + laneLo > q)      v0 = -1e30f;
                if (kb + 16 + laneLo > q) v1 = -1e30f;
            }
            float mx = fmaxf(v0, v1);
#pragma unroll
            for (int off = 1; off < 16; off <<= 1)
                mx = fmaxf(mx, __shfl_xor(mx, off, 32));
            const float mnew = fmaxf(mrow[r], mx);
            const float corr = __expf(mrow[r] - mnew);
            const float p0 = __expf(v0 - mnew);
            const float p1 = __expf(v1 - mnew);
            float ps = p0 + p1;
#pragma unroll
            for (int off = 1; off < 16; off <<= 1)
                ps += __shfl_xor(ps, off, 32);
            lrow[r] = lrow[r] * corr + ps;
            mrow[r] = mnew;
#pragma unroll
            for (int j = 0; j < 4; ++j) o[j][r] = o[j][r] * corr;
            const int row = r + 8 * hi;
            myLds[row * 32 + laneLo]      = f2bf(p0);
            myLds[row * 32 + 16 + laneLo] = f2bf(p1);
        }

        // Reload P (16x32) in WMMA A-fragment layout (same-wave LDS, in-order).
        const v16bf pA = load_a_frag(myLds + laneLo * 32, 0, hi);
#pragma unroll
        for (int j = 0; j < 4; ++j) {
            const unsigned short* vRow = Vt + ((size_t)bh * 64 + j * 16 + laneLo) * S;
            o[j] = wmma_bf16(pA, load_b_frag(vRow, kb, hi), o[j]);
        }
    }

#pragma unroll
    for (int r = 0; r < 8; ++r) {
        const float inv = 1.0f / lrow[r];
        const int q = qb + r + 8 * hi;
        unsigned short* orow = O + ((size_t)b * S + q) * D + h * 64;
#pragma unroll
        for (int j = 0; j < 4; ++j)
            orow[j * 16 + laneLo] = f2bf(o[j][r] * inv);
    }
}

// ---------------------------------------------------------------------------
// Host orchestration
// ---------------------------------------------------------------------------
extern "C" void kernel_launch(void* const* d_in, const int* in_sizes, int n_in,
                              void* d_out, int out_size, void* d_ws, size_t ws_size,
                              hipStream_t stream) {
    (void)in_sizes; (void)n_in; (void)out_size; (void)ws_size;
    constexpr int Bc = 4, Sc = 1024, Dc = 1024, Hc = 16, DFFc = 4096;
    const size_t tok = (size_t)Bc * Sc;   // 4096 rows
    const size_t MD  = tok * Dc;          // 4M elements

    char* ws = (char*)d_ws;
    size_t off = 0;
    auto alloc = [&](size_t bytes) -> void* {
        void* p = ws + off;
        off += (bytes + 255) & ~(size_t)255;
        return p;
    };

    unsigned short* w_saq = (unsigned short*)alloc((size_t)Dc * Dc * 2);
    unsigned short* w_sak = (unsigned short*)alloc((size_t)Dc * Dc * 2);
    unsigned short* w_sav = (unsigned short*)alloc((size_t)Dc * Dc * 2);
    unsigned short* w_sao = (unsigned short*)alloc((size_t)Dc * Dc * 2);
    unsigned short* w_caq = (unsigned short*)alloc((size_t)Dc * Dc * 2);
    unsigned short* w_cak = (unsigned short*)alloc((size_t)Dc * Dc * 2);
    unsigned short* w_cav = (unsigned short*)alloc((size_t)Dc * Dc * 2);
    unsigned short* w_cao = (unsigned short*)alloc((size_t)Dc * Dc * 2);
    unsigned short* w_ff1 = (unsigned short*)alloc((size_t)DFFc * Dc * 2);
    unsigned short* w_ff2 = (unsigned short*)alloc((size_t)Dc * DFFc * 2);
    unsigned short* prevb = (unsigned short*)alloc(MD * 2);
    unsigned short* hb    = (unsigned short*)alloc(MD * 2);
    unsigned short* qb_   = (unsigned short*)alloc(MD * 2);
    unsigned short* kb_   = (unsigned short*)alloc(MD * 2);
    unsigned short* vtb   = (unsigned short*)alloc(MD * 2);
    unsigned short* aob   = (unsigned short*)alloc(MD * 2);
    unsigned short* ffh   = (unsigned short*)alloc(tok * (size_t)DFFc * 2);

    float* y = (float*)d_out;  // residual accumulator lives in d_out (fp32)

    // y = this_seq
    (void)hipMemcpyAsync(y, d_in[0], MD * sizeof(float), hipMemcpyDeviceToDevice, stream);

    auto cast = [&](const void* src, unsigned short* dst, size_t n) {
        cast_bf16_kernel<<<dim3((unsigned)((n + 255) / 256)), dim3(256), 0, stream>>>(
            (const float*)src, dst, (int)n);
    };
    cast(d_in[3],  w_saq, (size_t)Dc * Dc);
    cast(d_in[4],  w_sak, (size_t)Dc * Dc);
    cast(d_in[5],  w_sav, (size_t)Dc * Dc);
    cast(d_in[6],  w_sao, (size_t)Dc * Dc);
    cast(d_in[7],  w_caq, (size_t)Dc * Dc);
    cast(d_in[8],  w_cak, (size_t)Dc * Dc);
    cast(d_in[9],  w_cav, (size_t)Dc * Dc);
    cast(d_in[10], w_cao, (size_t)Dc * Dc);
    cast(d_in[11], w_ff1, (size_t)DFFc * Dc);
    cast(d_in[13], w_ff2, (size_t)Dc * DFFc);
    cast(d_in[1],  prevb, MD);

    const dim3 blk(256);
    const dim3 gD(32, 8);             // M/128=32, N=1024/128=8
    const dim3 gFF1(32, 32);          // N=4096/128=32
    const dim3 gAttn(Bc * Hc, Sc / 128);

    // ---- Sublayer 1: masked self-attention -------------------------------
    ln_bf16_kernel<<<(unsigned)tok, blk, 0, stream>>>(
        y, (const float*)d_in[15], (const float*)d_in[16], hb, Dc);
    gemm_bf16_kernel<2, false, false><<<gD, blk, 0, stream>>>(
        hb, w_saq, nullptr, qb_, (int)tok, Dc, Dc, Sc, Hc);
    gemm_bf16_kernel<2, false, false><<<gD, blk, 0, stream>>>(
        hb, w_sak, nullptr, kb_, (int)tok, Dc, Dc, Sc, Hc);
    gemm_bf16_kernel<3, false, false><<<gD, blk, 0, stream>>>(
        hb, w_sav, nullptr, vtb, (int)tok, Dc, Dc, Sc, Hc);
    attn_kernel<<<gAttn, blk, 0, stream>>>(qb_, kb_, vtb, aob, Sc, Hc, 0.125f, 1);
    gemm_bf16_kernel<1, false, false><<<gD, blk, 0, stream>>>(
        aob, w_sao, nullptr, y, (int)tok, Dc, Dc, Sc, Hc);

    // ---- Sublayer 2: cross-attention (K/V from raw prev_seq) -------------
    ln_bf16_kernel<<<(unsigned)tok, blk, 0, stream>>>(
        y, (const float*)d_in[17], (const float*)d_in[18], hb, Dc);
    gemm_bf16_kernel<2, false, false><<<gD, blk, 0, stream>>>(
        hb, w_caq, nullptr, qb_, (int)tok, Dc, Dc, Sc, Hc);
    gemm_bf16_kernel<2, false, false><<<gD, blk, 0, stream>>>(
        prevb, w_cak, nullptr, kb_, (int)tok, Dc, Dc, Sc, Hc);
    gemm_bf16_kernel<3, false, false><<<gD, blk, 0, stream>>>(
        prevb, w_cav, nullptr, vtb, (int)tok, Dc, Dc, Sc, Hc);
    attn_kernel<<<gAttn, blk, 0, stream>>>(qb_, kb_, vtb, aob, Sc, Hc, 0.125f, 0);
    gemm_bf16_kernel<1, false, false><<<gD, blk, 0, stream>>>(
        aob, w_cao, nullptr, y, (int)tok, Dc, Dc, Sc, Hc);

    // ---- Sublayer 3: feed-forward ----------------------------------------
    ln_bf16_kernel<<<(unsigned)tok, blk, 0, stream>>>(
        y, (const float*)d_in[19], (const float*)d_in[20], hb, Dc);
    gemm_bf16_kernel<0, true, true><<<gFF1, blk, 0, stream>>>(
        hb, w_ff1, (const float*)d_in[12], ffh, (int)tok, DFFc, Dc, Sc, Hc);
    gemm_bf16_kernel<1, false, true><<<gD, blk, 0, stream>>>(
        ffh, w_ff2, (const float*)d_in[14], y, (int)tok, Dc, DFFc, Sc, Hc);
}
